// MHA_35003983462451
// MI455X (gfx1250) — compile-verified, matrix operand order checked by reading, not run on hardware
//
#include <hip/hip_runtime.h>

// ---------------------------------------------------------------------------
// MHA forward for MI455X (gfx1250, wave32, WMMA + TDM).
// ~69 GFLOP vs ~120MB HBM traffic => ~575 FLOP/B => compute bound, so all
// GEMMs run on v_wmma_f32_16x16x32_bf16 (bf16 operands, fp32 accumulate).
// Kernel 3 stages its A tile with TENSOR_LOAD_TO_LDS (TDM, TENSORcnt),
// using D# LDS padding to produce the bank-conflict-free 40-elem row stride.
// ---------------------------------------------------------------------------

typedef __attribute__((ext_vector_type(16))) __bf16 v16bf;
typedef __attribute__((ext_vector_type(8)))  __bf16 v8bf;
typedef __attribute__((ext_vector_type(4)))  __bf16 v4bf;
typedef __attribute__((ext_vector_type(8)))  float  v8f;
typedef __attribute__((ext_vector_type(4)))  unsigned int u32x4;
typedef __attribute__((ext_vector_type(8)))  int    i32x8;
typedef __attribute__((ext_vector_type(4)))  int    i32x4;

#ifndef __has_builtin
#define __has_builtin(x) 0
#endif
#if __has_builtin(__builtin_amdgcn_tensor_load_to_lds) && \
    __has_builtin(__builtin_amdgcn_s_wait_tensorcnt)
#define USE_TDM 1
#else
#define USE_TDM 0
#endif

namespace cfg {
constexpr int B  = 2;
constexpr int S  = 2048;
constexpr int D  = 1024;
constexpr int H  = 16;
constexpr int DK = 64;          // head dim
constexpr int N1 = 3 * D;       // 3072 qkv cols
constexpr int BM = 128;         // GEMM block tile M
constexpr int BN = 128;         // GEMM block tile N (8 wmma n-tiles / wave)
constexpr int BK = 32;          // GEMM k-step (one wmma K)
constexpr int NT = BN / 16;     // n-tiles per wave
constexpr int LP = 40;          // padded LDS row stride (bf16), 80B rows
constexpr int M  = B * S;       // 4096 tokens
}

// 16x32 A/B fragment for v_wmma_f32_16x16x32_bf16 (wave32):
// lane l holds row/col (l&15), k-base (l>>4)*8, two 16B runs at +0 / +16.
__device__ __forceinline__ v16bf load_frag(const __bf16* p) {
    v8bf lo = *(const v8bf*)(p);
    v8bf hi = *(const v8bf*)(p + 16);
    return __builtin_shufflevector(lo, hi, 0, 1, 2, 3, 4, 5, 6, 7,
                                           8, 9, 10, 11, 12, 13, 14, 15);
}

__device__ __forceinline__ v8f wmma_bf16(v16bf a, v16bf b, v8f c) {
    return __builtin_amdgcn_wmma_f32_16x16x32_bf16(
        false, a, false, b, (short)0, c, false, false);
}

// ---------------------------------------------------------------------------
// Kernel 1: qkv = x @ W_qkv + b_qkv   (fp32 in, bf16 Q / K / V^T out)
// grid = (M/128, 3072/128), block = 256 (8 waves x 16-row strips).
// ---------------------------------------------------------------------------
__global__ __launch_bounds__(256) void mha_qkv_gemm(
    const float* __restrict__ x, const float* __restrict__ Wqkv,
    const float* __restrict__ bqkv,
    __bf16* __restrict__ Qb, __bf16* __restrict__ Kb, __bf16* __restrict__ Vt)
{
    using namespace cfg;
    __shared__ __bf16 As[BM * LP];
    __shared__ __bf16 Bs[BN * LP];

    const int t    = threadIdx.x;
    const int wave = t >> 5;
    const int lane = t & 31;
    const int ml   = lane & 15;
    const int kb8  = (lane >> 4) * 8;
    const int m0   = blockIdx.x * BM;
    const int n0   = blockIdx.y * BN;

    v8f zf = {0.f, 0.f, 0.f, 0.f, 0.f, 0.f, 0.f, 0.f};
    v8f acc[NT];
    for (int nt = 0; nt < NT; ++nt) acc[nt] = zf;

    for (int k0 = 0; k0 < D; k0 += BK) {
        if (k0 + BK < D) {  // prefetch next k-tile (global_prefetch_b8)
            __builtin_prefetch(x + (size_t)(m0 + (t >> 3)) * D + (k0 + BK) + (t & 7) * 4, 0, 1);
            __builtin_prefetch(Wqkv + (size_t)(k0 + BK + (t >> 5)) * N1 + n0 + (t & 31) * 4, 0, 1);
        }
        // Stage A (128x32 fp32 -> bf16), float4-vectorized, coalesced.
        for (int i = 0; i < 4; ++i) {
            int e   = t + i * 256;           // 1024 float4 chunks
            int row = e >> 3;                // 8 float4 per 32-col row
            int c4  = e & 7;
            float4 v = *(const float4*)(x + (size_t)(m0 + row) * D + k0 + c4 * 4);
            v4bf pk = {(__bf16)v.x, (__bf16)v.y, (__bf16)v.z, (__bf16)v.w};
            *(v4bf*)&As[row * LP + c4 * 4] = pk;
        }
        // Stage B transposed: Bs[n][k] = W[k0+k][n0+n] (bf16).
        for (int i = 0; i < 4; ++i) {
            int e  = t + i * 256;            // 1024 float4 chunks
            int kk = e >> 5;                 // 32 float4 per 128-col k-row
            int c4 = e & 31;
            float4 v = *(const float4*)(Wqkv + (size_t)(k0 + kk) * N1 + n0 + c4 * 4);
            Bs[(c4 * 4 + 0) * LP + kk] = (__bf16)v.x;
            Bs[(c4 * 4 + 1) * LP + kk] = (__bf16)v.y;
            Bs[(c4 * 4 + 2) * LP + kk] = (__bf16)v.z;
            Bs[(c4 * 4 + 3) * LP + kk] = (__bf16)v.w;
        }
        __syncthreads();

        v16bf a = load_frag(&As[(wave * 16 + ml) * LP + kb8]);
        for (int nt = 0; nt < NT; ++nt) {
            v16bf b = load_frag(&Bs[(nt * 16 + ml) * LP + kb8]);
            acc[nt] = wmma_bf16(a, b, acc[nt]);
        }
        __syncthreads();
    }

    // Epilogue: bias + scatter. Region/head derived from the UNIFORM tile base
    // column (each 16-col tile sits wholly inside Q, K or V) => scalar branch.
    const int mrb = (lane >> 4) * 8;
    for (int nt = 0; nt < NT; ++nt) {
        int   c0   = n0 + nt * 16;           // uniform
        int   h    = c0 / (3 * DK);          // uniform (scalar div)
        int   r0   = c0 % (3 * DK);          // uniform
        float bias = bqkv[c0 + ml];
        size_t hb0 = (size_t)h;              // head within batch added below
        if (r0 < DK) {
            int r = r0 + ml;
            for (int i = 0; i < 8; ++i) {
                int tok = m0 + wave * 16 + mrb + i;
                int bb = tok >> 11, s = tok & (S - 1);
                Qb[(((size_t)bb * H + hb0) * S + s) * DK + r] =
                    (__bf16)(acc[nt][i] + bias);
            }
        } else if (r0 < 2 * DK) {
            int r = r0 - DK + ml;
            for (int i = 0; i < 8; ++i) {
                int tok = m0 + wave * 16 + mrb + i;
                int bb = tok >> 11, s = tok & (S - 1);
                Kb[(((size_t)bb * H + hb0) * S + s) * DK + r] =
                    (__bf16)(acc[nt][i] + bias);
            }
        } else {
            int r = r0 - 2 * DK + ml;
            for (int i = 0; i < 8; ++i) {
                int tok = m0 + wave * 16 + mrb + i;
                int bb = tok >> 11, s = tok & (S - 1);
                Vt[(((size_t)bb * H + hb0) * DK + r) * S + s] =
                    (__bf16)(acc[nt][i] + bias);
            }
        }
    }
}

// ---------------------------------------------------------------------------
// Kernel 2: flash attention. grid = (S/64, B*H), block = 128 (4 waves).
// Each wave owns 16 queries; streams 32-key blocks with online softmax.
// ---------------------------------------------------------------------------
__global__ __launch_bounds__(128) void mha_flash_attn(
    const __bf16* __restrict__ Qb, const __bf16* __restrict__ Kb,
    const __bf16* __restrict__ Vt, __bf16* __restrict__ Ob)
{
    using namespace cfg;
    __shared__ __bf16 Ps[4 * 16 * LP];       // per-wave 16x32 P tile (padded)

    const int t    = threadIdx.x;
    const int wave = t >> 5;
    const int lane = t & 31;
    const int ml   = lane & 15;
    const int kb8  = (lane >> 4) * 8;
    const int mrb  = (lane >> 4) * 8;

    const int bh = blockIdx.y;
    const int b  = bh >> 4;                  // / H
    const int h  = bh & 15;                  // % H
    const __bf16* Qp = Qb + (size_t)bh * S * DK;
    const __bf16* Kp = Kb + (size_t)bh * S * DK;
    const __bf16* Vp = Vt + (size_t)bh * DK * S;
    const int q0 = blockIdx.x * 64 + wave * 16;

    // Q fragments held in registers for the whole key sweep (dk 0..31, 32..63).
    v16bf qf0 = load_frag(Qp + (size_t)(q0 + ml) * DK + kb8);
    v16bf qf1 = load_frag(Qp + (size_t)(q0 + ml) * DK + 32 + kb8);

    v8f zf = {0.f, 0.f, 0.f, 0.f, 0.f, 0.f, 0.f, 0.f};
    v8f o[4] = {zf, zf, zf, zf};
    float mrow[8], lrow[8];
    for (int i = 0; i < 8; ++i) { mrow[i] = -3.0e38f; lrow[i] = 0.f; }

    __bf16* myP = &Ps[wave * 16 * LP];
    const float scale = 0.125f;              // 1/sqrt(64)

    for (int ks = 0; ks < S; ks += 32) {
        // S-tiles for 2x16 keys: S = Q @ K^T over dk=64 (2 wmma each).
        v8f sa0 = zf, sa1 = zf;
        {
            v16bf k0f = load_frag(Kp + (size_t)(ks + ml) * DK + kb8);
            sa0 = wmma_bf16(qf0, k0f, sa0);
            v16bf k1f = load_frag(Kp + (size_t)(ks + ml) * DK + 32 + kb8);
            sa0 = wmma_bf16(qf1, k1f, sa0);
            v16bf k2f = load_frag(Kp + (size_t)(ks + 16 + ml) * DK + kb8);
            sa1 = wmma_bf16(qf0, k2f, sa1);
            v16bf k3f = load_frag(Kp + (size_t)(ks + 16 + ml) * DK + 32 + kb8);
            sa1 = wmma_bf16(qf1, k3f, sa1);
        }

        // Online softmax over this 32-key block (rows across 16-lane groups).
        for (int i = 0; i < 8; ++i) {
            float s0 = sa0[i] * scale;
            float s1 = sa1[i] * scale;
            float rm = fmaxf(s0, s1);
            rm = fmaxf(rm, __shfl_xor(rm, 1, 32));
            rm = fmaxf(rm, __shfl_xor(rm, 2, 32));
            rm = fmaxf(rm, __shfl_xor(rm, 4, 32));
            rm = fmaxf(rm, __shfl_xor(rm, 8, 32));
            float mn   = fmaxf(mrow[i], rm);
            float corr = __expf(mrow[i] - mn);
            float p0   = __expf(s0 - mn);
            float p1   = __expf(s1 - mn);
            float rs   = p0 + p1;
            rs += __shfl_xor(rs, 1, 32);
            rs += __shfl_xor(rs, 2, 32);
            rs += __shfl_xor(rs, 4, 32);
            rs += __shfl_xor(rs, 8, 32);
            lrow[i] = lrow[i] * corr + rs;
            mrow[i] = mn;
            for (int nt = 0; nt < 4; ++nt) o[nt][i] *= corr;
            int mr = mrb + i;
            myP[mr * LP + ml]      = (__bf16)p0;   // D-layout -> [m][k] in LDS
            myP[mr * LP + 16 + ml] = (__bf16)p1;
        }

        // Wave-private LDS round-trip: order ds_store -> ds_load explicitly.
        asm volatile("s_wait_dscnt 0" ::: "memory");

        v16bf pf = load_frag(&myP[ml * LP + kb8]);
        for (int nt = 0; nt < 4; ++nt) {
            // B frag of V: col = dk (nt*16+ml), k-run contiguous in V^T.
            v16bf vf = load_frag(Vp + (size_t)(nt * 16 + ml) * S + ks + kb8);
            o[nt] = wmma_bf16(pf, vf, o[nt]);
        }
    }

    // Normalize and store attention output as bf16 [b][s][h*DK + dk].
    for (int i = 0; i < 8; ++i) {
        float inv = 1.0f / lrow[i];
        int   srow = q0 + mrb + i;
        size_t base = ((size_t)b * S + srow) * D + h * DK;
        for (int nt = 0; nt < 4; ++nt)
            Ob[base + nt * 16 + ml] = (__bf16)(o[nt][i] * inv);
    }
}

// ---------------------------------------------------------------------------
// Kernel 3: out = attn @ W_o + b_o  (bf16 A via TDM, fp32 out)
// grid = (M/128, 1024/128), block = 256.
// ---------------------------------------------------------------------------
__global__ __launch_bounds__(256) void mha_out_proj(
    const __bf16* __restrict__ A, const float* __restrict__ Wo,
    const float* __restrict__ bo, float* __restrict__ out)
{
    using namespace cfg;
    __shared__ __bf16 As[BM * LP];
    __shared__ __bf16 Bs[BN * LP];

    const int t    = threadIdx.x;
    const int wave = t >> 5;
    const int lane = t & 31;
    const int ml   = lane & 15;
    const int kb8  = (lane >> 4) * 8;
    const int m0   = blockIdx.x * BM;
    const int n0   = blockIdx.y * BN;

    v8f zf = {0.f, 0.f, 0.f, 0.f, 0.f, 0.f, 0.f, 0.f};
    v8f acc[NT];
    for (int nt = 0; nt < NT; ++nt) acc[nt] = zf;

    for (int k0 = 0; k0 < D; k0 += BK) {
        if (k0 + BK < D)
            __builtin_prefetch(Wo + (size_t)(k0 + BK + (t >> 5)) * D + n0 + (t & 31) * 4, 0, 1);

#if USE_TDM
        // Tensor Data Mover: DMA the 128x32 bf16 A tile into LDS, with D#
        // padding (16 DWORDs data + 4 DWORDs pad) producing the LP=40 layout.
        if (t < 32) {
            unsigned long long ga =
                (unsigned long long)(uintptr_t)(A + (size_t)m0 * D + k0);
            unsigned lds_off = (unsigned)(uintptr_t)(&As[0]);
            u32x4 g0 = {
                1u,                                    // count=1, no gather
                lds_off,                               // lds_addr (bytes)
                (unsigned)(ga & 0xffffffffu),          // global_addr[31:0]
                (unsigned)((ga >> 32) & 0x01ffffffu)   // global_addr[56:32]
                    | 0x80000000u                      // type=2 (image)
            };
            i32x8 g1 = {
                (int)((1u << 16) |                     // data_size = 2B
                      (1u << 20) |                     // pad_enable
                      (3u << 22) |                     // pad_interval: 16 DW
                      (3u << 25)),                     // pad_amount: 4 DW
                (int)(((unsigned)D & 0xffffu) << 16),  // tensor_dim0 lo16
                (int)((((unsigned)D >> 16) & 0xffffu)  // tensor_dim0 hi16
                      | (((unsigned)M & 0xffffu) << 16)),  // tensor_dim1 lo16
                (int)((((unsigned)M >> 16) & 0xffffu)
                      | (32u << 16)),                  // tile_dim0 = 32
                128,                                   // tile_dim1 = 128
                (int)D,                                // dim0_stride lo32
                0, 0
            };
            i32x4 z4 = {0, 0, 0, 0};
#if __has_include(<hip/amd_detail/amd_gfx1250_TDM.h>)
            i32x8 z8 = {0, 0, 0, 0, 0, 0, 0, 0};
            __builtin_amdgcn_tensor_load_to_lds(g0, g1, z4, z4, z8, 0);
#else
            __builtin_amdgcn_tensor_load_to_lds(g0, g1, z4, z4, 0);
#endif
        }
#else
        for (int i = 0; i < 2; ++i) {
            int e   = t + i * 256;
            int row = e >> 2;
            int c8  = e & 3;
            v8bf v = *(const v8bf*)(A + (size_t)(m0 + row) * D + k0 + c8 * 8);
            *(v8bf*)&As[row * LP + c8 * 8] = v;
        }
#endif
        // Stage B transposed (overlaps with the TDM transfer).
        for (int i = 0; i < 4; ++i) {
            int e  = t + i * 256;
            int kk = e >> 5;
            int c4 = e & 31;
            float4 v = *(const float4*)(Wo + (size_t)(k0 + kk) * D + n0 + c4 * 4);
            Bs[(c4 * 4 + 0) * LP + kk] = (__bf16)v.x;
            Bs[(c4 * 4 + 1) * LP + kk] = (__bf16)v.y;
            Bs[(c4 * 4 + 2) * LP + kk] = (__bf16)v.z;
            Bs[(c4 * 4 + 3) * LP + kk] = (__bf16)v.w;
        }
#if USE_TDM
        if (t < 32) __builtin_amdgcn_s_wait_tensorcnt((short)0);
#endif
        __syncthreads();

        v16bf a = load_frag(&As[(wave * 16 + ml) * LP + kb8]);
        for (int nt = 0; nt < NT; ++nt) {
            v16bf b = load_frag(&Bs[(nt * 16 + ml) * LP + kb8]);
            acc[nt] = wmma_bf16(a, b, acc[nt]);
        }
        __syncthreads();
    }

    const int mrb = (lane >> 4) * 8;
    for (int nt = 0; nt < NT; ++nt) {
        int   c    = n0 + nt * 16 + ml;
        float bias = bo[c];
        for (int i = 0; i < 8; ++i) {
            int m = wave * 16 + mrb + i;
            out[(size_t)(m0 + m) * D + c] = acc[nt][i] + bias;
        }
    }
}

// ---------------------------------------------------------------------------
extern "C" void kernel_launch(void* const* d_in, const int* in_sizes, int n_in,
                              void* d_out, int out_size, void* d_ws, size_t ws_size,
                              hipStream_t stream) {
    using namespace cfg;
    (void)in_sizes; (void)n_in; (void)out_size; (void)ws_size;

    const float* x    = (const float*)d_in[0];
    const float* Wqkv = (const float*)d_in[1];
    const float* bqkv = (const float*)d_in[2];
    const float* Wo   = (const float*)d_in[3];
    const float* bo   = (const float*)d_in[4];
    float*       out  = (float*)d_out;

    // Workspace: Q | K | V^T | attn-out, each B*H*S*DK bf16 (8MB) = 32MB.
    const size_t per = (size_t)B * H * S * DK;
    __bf16* Qb = (__bf16*)d_ws;
    __bf16* Kb = Qb + per;
    __bf16* Vt = Kb + per;
    __bf16* Ob = Vt + per;

    mha_qkv_gemm<<<dim3(M / BM, N1 / BN), 256, 0, stream>>>(
        x, Wqkv, bqkv, Qb, Kb, Vt);
    mha_flash_attn<<<dim3(S / 64, B * H), 128, 0, stream>>>(
        Qb, Kb, Vt, Ob);
    mha_out_proj<<<dim3(M / BM, D / BN), 256, 0, stream>>>(
        Ob, Wo, bo, out);
}
